// MCILatencyModel_40389872452040
// MI455X (gfx1250) — matrix operand order repaired and checked by reading, not compiled, not run on hardware
//
#include <hip/hip_runtime.h>
#include <math.h>

// ---------------------------------------------------------------------------
// GNN latency model for MI455X (gfx1250, wave32).
//  - Node GEMMs via v_wmma_f32_16x16x32_f16 (16x16 tiles, K-step 32)
//  - Edge softmax/aggregation via two fused atomic passes (L2-resident)
//  - BN folded to per-channel affine; LN/ReLU/residual fused per node
// ---------------------------------------------------------------------------

#define EPS 1e-5f
#define NEG_SLOPE 0.2f
#define NN 50000          // nodes
#define NB 512            // graphs
#define HID 128
#define HEADS 8
#define HC 16             // head dim

typedef __attribute__((ext_vector_type(8)))  _Float16 v8h;
typedef __attribute__((ext_vector_type(16))) _Float16 v16h;
typedef __attribute__((ext_vector_type(8)))  float    v8f;

// ---------------- small utility kernels ----------------

__global__ void fill_kernel(float* __restrict__ p, float v, int n) {
  int i = blockIdx.x * blockDim.x + threadIdx.x;
  if (i < n) p[i] = v;
}

__global__ void cvt_f16_kernel(const float* __restrict__ x, _Float16* __restrict__ y, int n) {
  int i = blockIdx.x * blockDim.x + threadIdx.x;
  if (i < n) y[i] = (_Float16)x[i];
}

// W [K,M] f32 -> Wt [M,K] f16 (row m of Wt = column m of W)
__global__ void transpose_cvt_kernel(const float* __restrict__ W, _Float16* __restrict__ Wt,
                                     int K, int M) {
  int i = blockIdx.x * blockDim.x + threadIdx.x;
  if (i >= K * M) return;
  int k = i / M, m = i - k * M;
  Wt[m * K + k] = (_Float16)W[k * M + m];
}

// Fold eval-mode BN (+ preceding bias) into per-channel affine: y = x*s + t
__global__ void fold_bn_kernel(const float* __restrict__ b, const float* __restrict__ rm,
                               const float* __restrict__ rv, const float* __restrict__ g,
                               const float* __restrict__ bb,
                               float* __restrict__ s, float* __restrict__ t, int d) {
  int c = blockIdx.x * blockDim.x + threadIdx.x;
  if (c >= d) return;
  float inv = rsqrtf(rv[c] + EPS);
  float sc  = g[c] * inv;
  s[c] = sc;
  t[c] = ((b ? b[c] : 0.f) - rm[c]) * sc + bb[c];
}

// ---------------- WMMA GEMM:  Y[N,128] = X16[N,K] @ W (Wt16 = W^T, [128,K]) --
// Block = 256 threads = 8 waves; block handles a 16-row strip, wave w handles
// output columns [16w, 16w+16). K stepped by 32 per v_wmma_f32_16x16x32_f16.
// Optional fused epilogue: y = relu(y*sc[col] + sh[col]).
__global__ __launch_bounds__(256)
void gemm_wmma_kernel(const _Float16* __restrict__ X16, const _Float16* __restrict__ Wt16,
                      float* __restrict__ Y, int N, int K,
                      const float* __restrict__ sc, const float* __restrict__ sh, int relu) {
  const int wave = threadIdx.x >> 5;
  const int lane = threadIdx.x & 31;
  const int hi   = lane >> 4;        // half-wave select
  const int l    = lane & 15;
  const int row0 = blockIdx.x * 16;
  const int col0 = wave * 16;
  const int arow = row0 + l;         // A row owned by this lane
  const int bcol = col0 + l;         // B column owned by this lane

  v8f acc = {};
  const _Float16* arowp = X16 + (size_t)arow * K;
  const _Float16* bcolp = Wt16 + (size_t)bcol * K;

  for (int kk = 0; kk < K; kk += 32) {
    // A 16x32 f16: lane<16 -> K in {kk..kk+7, kk+16..kk+23}; lane>=16 -> +8
    int kA = kk + hi * 8;
    v8h a0 = *(const v8h*)(arowp + kA);
    v8h a1 = *(const v8h*)(arowp + kA + 16);
    v16h a = __builtin_shufflevector(a0, a1, 0,1,2,3,4,5,6,7,8,9,10,11,12,13,14,15);
    // B 32x16 f16: lane<16 -> K in {kk..kk+7, kk+16..kk+23}; lane>=16 -> +8
    int kB = kk + hi * 8;
    v8h b0 = *(const v8h*)(bcolp + kB);
    v8h b1 = *(const v8h*)(bcolp + kB + 16);
    v16h b = __builtin_shufflevector(b0, b1, 0,1,2,3,4,5,6,7,8,9,10,11,12,13,14,15);
    acc = __builtin_amdgcn_wmma_f32_16x16x32_f16(false, a, false, b,
                                                 (short)0, acc, false, false);
  }
  // D: VGPR r -> row = row0 + r + 8*hi, col = col0 + (lane&15)
  const int ocol = col0 + l;
  float s = sc ? sc[ocol] : 1.f;
  float t = sc ? sh[ocol] : 0.f;
  #pragma unroll
  for (int r = 0; r < 8; ++r) {
    int orow = row0 + r + hi * 8;
    float v = acc[r];
    if (sc)  v = v * s + t;
    if (relu) v = fmaxf(v, 0.f);
    if (orow < N) Y[(size_t)orow * HID + ocol] = v;
  }
}

// ---------------- GAT attention scores: a_src/a_dst [N,8] --------------------
__global__ void attn_scores_kernel(const float* __restrict__ xp,
                                   const float* __restrict__ att_src,
                                   const float* __restrict__ att_dst,
                                   float* __restrict__ as, float* __restrict__ ad, int N) {
  int i = blockIdx.x * blockDim.x + threadIdx.x;   // n*8 + h
  if (i >= N * HEADS) return;
  int n = i >> 3, h = i & 7;
  const float* v = xp + (size_t)n * HID + h * HC;
  const float* s = att_src + h * HC;
  const float* d = att_dst + h * HC;
  float a = 0.f, b = 0.f;
  #pragma unroll
  for (int c = 0; c < HC; ++c) { a += v[c] * s[c]; b += v[c] * d[c]; }
  as[i] = a; ad[i] = b;
}

// float atomic max via ordered-int trick (init must be -inf)
__device__ __forceinline__ void atomicMaxF(float* addr, float v) {
  if (v >= 0.f) atomicMax((int*)addr, __float_as_int(v));
  else          atomicMin((unsigned int*)addr, (unsigned int)__float_as_int(v));
}

__device__ __forceinline__ float leaky(float x) { return x > 0.f ? x : NEG_SLOPE * x; }

// Pass 1: per-edge logits -> segment max over dst. tid = edge (E real + N self)
__global__ void edge_max_kernel(const int* __restrict__ src, const int* __restrict__ dst,
                                const float* __restrict__ as, const float* __restrict__ ad,
                                float* __restrict__ m, int E, int N) {
  int i = blockIdx.x * blockDim.x + threadIdx.x;
  if (i >= E + N) return;
  int s, d;
  if (i < E) { s = src[i]; d = dst[i]; } else { s = i - E; d = s; }
  const float4* a0 = (const float4*)(as + (size_t)s * HEADS);
  const float4* b0 = (const float4*)(ad + (size_t)d * HEADS);
  float4 a[2] = { a0[0], a0[1] }, b[2] = { b0[0], b0[1] };
  const float* af = (const float*)a; const float* bf = (const float*)b;
  #pragma unroll
  for (int h = 0; h < HEADS; ++h)
    atomicMaxF(&m[(size_t)d * HEADS + h], leaky(af[h] + bf[h]));
}

// Pass 2: w = exp(e-m); accumulate ssum[dst,h] += w; agg[dst,h,:] += w*xp[src,h,:]
// tid = (edge, head)
__global__ void edge_agg_kernel(const int* __restrict__ src, const int* __restrict__ dst,
                                const float* __restrict__ as, const float* __restrict__ ad,
                                const float* __restrict__ m, float* __restrict__ ssum,
                                const float* __restrict__ xp, float* __restrict__ agg,
                                int E, int N) {
  int i = blockIdx.x * blockDim.x + threadIdx.x;
  int total = (E + N) * HEADS;
  if (i >= total) return;
  int e = i >> 3, h = i & 7;
  int s, d;
  if (e < E) { s = src[e]; d = dst[e]; } else { s = e - E; d = s; }
  float ev = leaky(as[(size_t)s * HEADS + h] + ad[(size_t)d * HEADS + h]);
  float w = __expf(ev - m[(size_t)d * HEADS + h]);
  atomicAdd(&ssum[(size_t)d * HEADS + h], w);
  const float* xs = xp + (size_t)s * HID + h * HC;
  float* ag = agg + (size_t)d * HID + h * HC;
  #pragma unroll
  for (int c = 0; c < HC; ++c) atomicAdd(&ag[c], w * xs[c]);
}

// Finalize (concat layers, D=128): normalize, BN-affine, LN, ReLU, +residual.
// In-place safe on agg (per-node block reads, syncs, then writes).
__global__ __launch_bounds__(128)
void gat_finalize_concat_kernel(const float* __restrict__ agg, const float* __restrict__ ssum,
                                const float* __restrict__ bn_s, const float* __restrict__ bn_t,
                                const float* __restrict__ ln_g, const float* __restrict__ ln_b,
                                const float* __restrict__ hres, float* __restrict__ out) {
  int n = blockIdx.x, c = threadIdx.x;
  __shared__ float r1[128], r2[128];
  float v = agg[(size_t)n * HID + c] / ssum[(size_t)n * HEADS + (c >> 4)];
  float u = v * bn_s[c] + bn_t[c];
  r1[c] = u; r2[c] = u * u;
  __syncthreads();
  for (int st = 64; st > 0; st >>= 1) {
    if (c < st) { r1[c] += r1[c + st]; r2[c] += r2[c + st]; }
    __syncthreads();
  }
  float mu = r1[0] * (1.f / 128.f);
  float var = r2[0] * (1.f / 128.f) - mu * mu;
  float z = (u - mu) * rsqrtf(var + EPS) * ln_g[c] + ln_b[c];
  z = fmaxf(z, 0.f) + hres[(size_t)n * HID + c];
  out[(size_t)n * HID + c] = z;
}

// Finalize (last layer, concat=False): mean over heads -> D=16; BN, LN, ReLU.
__global__ __launch_bounds__(32)
void gat_finalize_mean_kernel(const float* __restrict__ agg, const float* __restrict__ ssum,
                              const float* __restrict__ bn_s, const float* __restrict__ bn_t,
                              const float* __restrict__ ln_g, const float* __restrict__ ln_b,
                              float* __restrict__ out) {
  int n = blockIdx.x, c = threadIdx.x;
  float u = 0.f;
  if (c < HC) {
    float acc = 0.f;
    #pragma unroll
    for (int h = 0; h < HEADS; ++h)
      acc += agg[(size_t)n * HID + h * HC + c] / ssum[(size_t)n * HEADS + h];
    float v = acc * 0.125f;
    u = v * bn_s[c] + bn_t[c];
  }
  float s1 = u, s2 = u * u;
  #pragma unroll
  for (int mk = 8; mk > 0; mk >>= 1) {
    s1 += __shfl_xor(s1, mk, 32);
    s2 += __shfl_xor(s2, mk, 32);
  }
  float mu = s1 * (1.f / 16.f), var = s2 * (1.f / 16.f) - mu * mu;
  if (c < HC) {
    float z = (u - mu) * rsqrtf(var + EPS) * ln_g[c] + ln_b[c];
    out[(size_t)n * HC + c] = fmaxf(z, 0.f);
  }
}

// Global mean/max pool over contiguous sorted batch segments -> hp [512,32]
__global__ __launch_bounds__(32)
void pool_kernel(const float* __restrict__ h, float* __restrict__ hp) {
  int g = blockIdx.x, c = threadIdx.x;
  if (c >= HC) return;
  int s = (g * NN + NB - 1) >> 9;
  int e = ((g + 1) * NN + NB - 1) >> 9;
  float sum = 0.f, mx = -3.4e38f;
  for (int i = s; i < e; ++i) {
    float v = h[(size_t)i * HC + c];
    sum += v; mx = fmaxf(mx, v);
  }
  float cnt = (float)(e - s);
  hp[g * 32 + c]      = sum / fmaxf(cnt, 1.f);
  hp[g * 32 + HC + c] = mx;
}

// Fused dense: Y = [relu]([LN](X@W * bn_s + bn_t)). One block per row.
// If extra != null, a per-row scalar is appended to X (K = xcols + 1).
__global__ __launch_bounds__(256)
void dense_kernel(const float* __restrict__ X, const float* __restrict__ extra,
                  const float* __restrict__ W,
                  const float* __restrict__ bn_s, const float* __restrict__ bn_t,
                  const float* __restrict__ ln_g, const float* __restrict__ ln_b,
                  int relu, float* __restrict__ Y, int K, int O) {
  extern __shared__ float sm[];
  float* xs = sm; float* ys = sm + K;
  int b = blockIdx.x, t = threadIdx.x;
  int nx = extra ? K - 1 : K;
  for (int i = t; i < nx; i += 256) xs[i] = X[(size_t)b * nx + i];
  if (extra && t == 0) xs[K - 1] = extra[b];
  __syncthreads();
  for (int o = t; o < O; o += 256) {
    float acc = 0.f;
    for (int k = 0; k < K; ++k) acc += xs[k] * W[(size_t)k * O + o];
    ys[o] = acc * bn_s[o] + bn_t[o];
  }
  __syncthreads();
  if (ln_g) {
    __shared__ float r1[256], r2[256];
    float a1 = 0.f, a2 = 0.f;
    for (int o = t; o < O; o += 256) { a1 += ys[o]; a2 += ys[o] * ys[o]; }
    r1[t] = a1; r2[t] = a2;
    __syncthreads();
    for (int st = 128; st > 0; st >>= 1) {
      if (t < st) { r1[t] += r1[t + st]; r2[t] += r2[t + st]; }
      __syncthreads();
    }
    float mu = r1[0] / (float)O, var = r2[0] / (float)O - mu * mu;
    float inv = rsqrtf(var + EPS);
    for (int o = t; o < O; o += 256) {
      float z = (ys[o] - mu) * inv * ln_g[o] + ln_b[o];
      if (relu) z = fmaxf(z, 0.f);
      Y[(size_t)b * O + o] = z;
    }
  } else {
    for (int o = t; o < O; o += 256) {
      float z = ys[o];
      if (relu) z = fmaxf(z, 0.f);
      Y[(size_t)b * O + o] = z;
    }
  }
}

__global__ void head_kernel(const float* __restrict__ X, const float* __restrict__ W,
                            const float* __restrict__ b, float* __restrict__ lat) {
  int r = blockIdx.x * blockDim.x + threadIdx.x;
  if (r >= NB) return;
  float acc = b[0];
  #pragma unroll 4
  for (int k = 0; k < 128; ++k) acc += X[(size_t)r * 128 + k] * W[k];
  lat[r] = acc;
}

// ---------------------------------------------------------------------------

extern "C" void kernel_launch(void* const* d_in, const int* in_sizes, int n_in,
                              void* d_out, int out_size, void* d_ws, size_t ws_size,
                              hipStream_t stream) {
  // Input order: x, dop, <params leaves (JAX sorted-key pytree)>, edge_index, batch
  const float* x_in  = (const float*)d_in[0];
  const float* dop   = (const float*)d_in[1];
  // params leaf indices (sorted keys: ep, ep_bn, ep_ln, gat, gp, gp_bn, head, inp, inp_bn, mlp)
  const int EP_W = 2, EP_B = 3, EPBN = 4 /* bb,g,rm,rv */, EPLN = 8;
  const int GAT0 = 12;               // per layer: W,att_dst,att_src,bias,bn{bb,g,rm,rv},ln{bb,g,rm,rv}
  const int GP_W = 48, GP_B = 49, GPBN = 50;
  const int HD_W = 54, HD_B = 55;
  const int IN_W = 56, IN_B = 57, INBN = 58;
  const int MLP0 = 62;               // per layer: W,b,bn{bb,g,rm,rv},ln{bb,g,rm,rv}
  const int* edge = (const int*)d_in[92];
  const int E = in_sizes[92] >> 1;
  const int* esrc = edge;
  const int* edst = edge + E;

  float* lat_out = (float*)d_out;          // [512]
  float* emb_out = (float*)d_out + NB;     // [512,256]

  // ---- workspace carve-up (256B aligned) ----
  char* base = (char*)d_ws;
  size_t off = 0;
  auto carve = [&](size_t bytes) -> char* {
    char* p = base + off;
    off = (off + bytes + 255) & ~(size_t)255;
    return p;
  };
  float*    bufA  = (float*)   carve((size_t)NN * HID * 4);
  float*    bufB  = (float*)   carve((size_t)NN * HID * 4);
  float*    bufC  = (float*)   carve((size_t)NN * HID * 4);
  _Float16* h16   = (_Float16*)carve((size_t)NN * HID * 2);
  _Float16* wt16  = (_Float16*)carve((size_t)HID * HID * 2);
  float*    as    = (float*)   carve((size_t)NN * HEADS * 4);
  float*    ad    = (float*)   carve((size_t)NN * HEADS * 4);
  float*    mmax  = (float*)   carve((size_t)NN * HEADS * 4);
  float*    ssum  = (float*)   carve((size_t)NN * HEADS * 4);
  float*    hsml  = (float*)   carve((size_t)NN * HC * 4);
  float*    hp    = (float*)   carve((size_t)NB * 32 * 4);
  float*    t1    = (float*)   carve((size_t)NB * 128 * 4);
  float*    t2    = (float*)   carve((size_t)NB * 512 * 4);
  float*    t3    = (float*)   carve((size_t)NB * 256 * 4);
  float*    t4    = (float*)   carve((size_t)NB * 128 * 4);
  float*    fs    = (float*)   carve(512 * 4);
  float*    ft    = (float*)   carve(512 * 4);
  (void)ws_size; (void)n_in; (void)out_size;

  auto P = [&](int i) { return (const float*)d_in[i]; };
  const int TPB = 256;
  auto gr = [&](long n) { return (int)((n + TPB - 1) / TPB); };

  // ================= input projection: relu(BN(x@W + b)) =================
  cvt_f16_kernel<<<gr((long)NN * 64), TPB, 0, stream>>>(x_in, h16, NN * 64);
  transpose_cvt_kernel<<<gr(64 * 128), TPB, 0, stream>>>(P(IN_W), wt16, 64, 128);
  fold_bn_kernel<<<1, 128, 0, stream>>>(P(IN_B), P(INBN + 2), P(INBN + 3),
                                        P(INBN + 1), P(INBN + 0), fs, ft, 128);
  gemm_wmma_kernel<<<NN / 16, 256, 0, stream>>>(h16, wt16, bufA, NN, 64, fs, ft, 1);

  // ================= 3 GAT layers =================
  float* rot[3] = { bufA, bufB, bufC };   // h_cur, xp, agg rotate
  float* h_cur = bufA;
  for (int L = 0; L < 3; ++L) {
    const int bq = GAT0 + 12 * L;
    const float* gW   = P(bq + 0);
    const float* attd = P(bq + 1);
    const float* atts = P(bq + 2);
    const float* gbias= P(bq + 3);
    const float* bn_bb= P(bq + 4), *bn_g = P(bq + 5), *bn_rm = P(bq + 6), *bn_rv = P(bq + 7);
    const float* ln_bb= P(bq + 8), *ln_g = P(bq + 9);
    float* xp  = bufB;
    float* agg = (L == 0) ? bufC : (L == 1) ? bufA : bufC;

    cvt_f16_kernel<<<gr((long)NN * HID), TPB, 0, stream>>>(h_cur, h16, NN * HID);
    transpose_cvt_kernel<<<gr(HID * HID), TPB, 0, stream>>>(gW, wt16, HID, HID);
    gemm_wmma_kernel<<<NN / 16, 256, 0, stream>>>(h16, wt16, xp, NN, HID,
                                                  nullptr, nullptr, 0);
    attn_scores_kernel<<<gr((long)NN * HEADS), TPB, 0, stream>>>(xp, atts, attd, as, ad, NN);

    fill_kernel<<<gr((long)NN * HEADS), TPB, 0, stream>>>(mmax, -INFINITY, NN * HEADS);
    fill_kernel<<<gr((long)NN * HEADS), TPB, 0, stream>>>(ssum, 0.f, NN * HEADS);
    fill_kernel<<<gr((long)NN * HID),   TPB, 0, stream>>>(agg,  0.f, NN * HID);

    edge_max_kernel<<<gr((long)E + NN), TPB, 0, stream>>>(esrc, edst, as, ad, mmax, E, NN);
    edge_agg_kernel<<<gr(((long)E + NN) * HEADS), TPB, 0, stream>>>(
        esrc, edst, as, ad, mmax, ssum, xp, agg, E, NN);

    int d = (L < 2) ? HID : HC;
    fold_bn_kernel<<<1, 128, 0, stream>>>(gbias, bn_rm, bn_rv, bn_g, bn_bb, fs, ft, d);
    if (L < 2) {
      gat_finalize_concat_kernel<<<NN, 128, 0, stream>>>(agg, ssum, fs, ft,
                                                         ln_g, ln_bb, h_cur, agg);
      h_cur = agg;            // L0: h=bufC ; L1: h=bufA
    } else {
      gat_finalize_mean_kernel<<<NN, 32, 0, stream>>>(agg, ssum, fs, ft, ln_g, ln_bb, hsml);
    }
  }

  // ================= pooling + dense tail =================
  pool_kernel<<<NB, 32, 0, stream>>>(hsml, hp);

  // gp: [32 -> 128], BN, ReLU
  fold_bn_kernel<<<1, 128, 0, stream>>>(P(GP_B), P(GPBN + 2), P(GPBN + 3),
                                        P(GPBN + 1), P(GPBN + 0), fs, ft, 128);
  dense_kernel<<<NB, 256, (32 + 128) * 4, stream>>>(hp, nullptr, P(GP_W), fs, ft,
                                                    nullptr, nullptr, 1, t1, 32, 128);
  // ep: [128 -> 256], BN, LN, ReLU -> emb (written straight into d_out)
  fold_bn_kernel<<<1, 256, 0, stream>>>(P(EP_B), P(EPBN + 2), P(EPBN + 3),
                                        P(EPBN + 1), P(EPBN + 0), fs, ft, 256);
  dense_kernel<<<NB, 256, (128 + 256) * 4, stream>>>(t1, nullptr, P(EP_W), fs, ft,
                                                     P(EPLN + 1), P(EPLN + 0), 1,
                                                     emb_out, 128, 256);
  // mlp: [emb|dop] 257 -> 512 -> 256 -> 128 (BN, LN, ReLU each)
  const int mO[3] = { 512, 256, 128 };
  const float* mX[3] = { emb_out, t2, t3 };
  float*       mY[3] = { t2, t3, t4 };
  const int mK[3] = { 257, 512, 256 };
  for (int j = 0; j < 3; ++j) {
    const int mb = MLP0 + 10 * j;
    fold_bn_kernel<<<(mO[j] + 127) / 128, 128, 0, stream>>>(
        P(mb + 1), P(mb + 4), P(mb + 5), P(mb + 3), P(mb + 2), fs, ft, mO[j]);
    dense_kernel<<<NB, 256, (size_t)(mK[j] + mO[j]) * 4, stream>>>(
        mX[j], (j == 0) ? dop : nullptr, P(mb + 0), fs, ft,
        P(mb + 7), P(mb + 6), 1, mY[j], mK[j], mO[j]);
  }
  // head: [128 -> 1]
  head_kernel<<<2, 256, 0, stream>>>(t4, P(HD_W), P(HD_B), lat_out);
}
